// SelfAttention_89335319757388
// MI455X (gfx1250) — matrix operand order repaired and checked by reading, not compiled, not run on hardware
//
#include <hip/hip_runtime.h>

#define B_ 8
#define C_ 512
#define L_ 2048

typedef __bf16 bf16_t;
typedef bf16_t v16bf __attribute__((ext_vector_type(16)));
typedef float  v8f   __attribute__((ext_vector_type(8)));

// ---------- bf16 helpers ----------

__device__ __forceinline__ unsigned short f2bf(float f) {
  unsigned u = __float_as_uint(f);
  u += 0x7fffu + ((u >> 16) & 1u);           // round-to-nearest-even
  return (unsigned short)(u >> 16);
}
__device__ __forceinline__ unsigned pack2(float lo, float hi) {
  return (unsigned)f2bf(lo) | ((unsigned)f2bf(hi) << 16);
}

union Frag { v16bf v; unsigned u[8]; };

// A-matrix 16x32 bf16 fragment: lane m = lane&15, VGPR j holds
//   K = half*8 + 2j      (j<4)
//   K = 8 + half*8 + 2j  (j>=4)
// Tile stored [M][K] K-contiguous -> packed dword loads.
__device__ __forceinline__ v16bf load_a(const unsigned short* row, int kbase, int half) {
  Frag r;
#pragma unroll
  for (int j = 0; j < 8; ++j) {
    int k = kbase + ((j < 4) ? (half * 8 + 2 * j) : (8 + half * 8 + 2 * j));
    r.u[j] = *(const unsigned*)(row + k);
  }
  return r.v;
}
// B-matrix 32x16 bf16 fragment: lane n = lane&15, VGPR j holds K = half*8 + 2j (j<4), 8 + half*8 + 2j (j>=4).
// Tile stored [N][K] K-contiguous -> packed dword loads.
__device__ __forceinline__ v16bf load_b(const unsigned short* row, int kbase, int half) {
  Frag r;
#pragma unroll
  for (int j = 0; j < 8; ++j)
    r.u[j] = *(const unsigned*)(row + kbase + ((j < 4) ? (half * 8 + 2 * j) : (8 + half * 8 + 2 * j)));
  return r.v;
}
__device__ __forceinline__ v8f wmma_bf16(v16bf a, v16bf b, v8f c) {
  return __builtin_amdgcn_wmma_f32_16x16x32_bf16(false, a, false, b, (short)0, c,
                                                 false, false);
}

// ---------- async global->LDS staging (gfx1250 GLOBAL_LOAD_ASYNC_TO_LDS_B128) ----------
// Builtin signature (from compiler diagnostic): (int4 AS1* src, int4 AS3* dst, Ii, Ii).

#if __has_builtin(__builtin_amdgcn_global_load_async_to_lds_b128)
#define HAVE_ASYNC_LDS 1
typedef int v4i_t __attribute__((ext_vector_type(4)));
typedef __attribute__((address_space(1))) v4i_t g_v4i;
typedef __attribute__((address_space(3))) v4i_t l_v4i;
__device__ __forceinline__ void async_copy16(void* lds, const void* g) {
  // Generic LDS pointers carry the LDS byte offset in bits [31:0]; global
  // addresses are flat addresses -> integer reinterpretation is exact.
  __builtin_amdgcn_global_load_async_to_lds_b128(
      (g_v4i*)(uintptr_t)g, (l_v4i*)(unsigned)(uintptr_t)lds, 0, 0);
}
__device__ __forceinline__ void async_wait0() {
#if __has_builtin(__builtin_amdgcn_s_wait_asynccnt)
  __builtin_amdgcn_s_wait_asynccnt(0);
#else
  asm volatile("s_wait_asynccnt 0" ::: "memory");
#endif
}
#else
#define HAVE_ASYNC_LDS 0
#endif

// Copy n16*16 bytes per thread from global to LDS (both 16B aligned).
__device__ __forceinline__ void stage16(unsigned short* lds, const unsigned short* g,
                                        int n16) {
#if HAVE_ASYNC_LDS
  for (int k = 0; k < n16; ++k) async_copy16(lds + k * 8, g + k * 8);
#else
  const uint4* gs = (const uint4*)g;
  uint4* d = (uint4*)lds;
  for (int k = 0; k < n16; ++k) d[k] = gs[k];
#endif
}
__device__ __forceinline__ void stage_fence() {
#if HAVE_ASYNC_LDS
  async_wait0();
#endif
}

// ============================================================
// Kernel 1: fused QKV projection (conv1d k=1 == GEMM) + bias,
// outputs bf16 in attention-friendly layouts:
//   Q,K : [b][l][co]   (K-contiguous for score-GEMM fragments)
//   V   : [b][co][m]   (K-contiguous for PV-GEMM B fragments)
// grid: (L/128, C/64, B*3); block 256 (8 waves), wave tile 32x32.
// ============================================================
__global__ __launch_bounds__(256)
void qkv_proj_kernel(const float* __restrict__ x,
                     const float* __restrict__ Wq, const float* __restrict__ bq,
                     const float* __restrict__ Wk, const float* __restrict__ bk,
                     const float* __restrict__ Wv, const float* __restrict__ bv,
                     unsigned short* __restrict__ Qb,
                     unsigned short* __restrict__ Kb,
                     unsigned short* __restrict__ Vb) {
  __shared__ unsigned short ws[64 * 40];    // W tile  [co][ci]  (+8 pad)
  __shared__ unsigned short xs[128 * 40];   // x tile  [l][ci]   (+8 pad, transposed)

  const int tid = threadIdx.x;
  const int lane = tid & 31, w = tid >> 5;
  const int half = lane >> 4, ln = lane & 15;
  const int wrow = w >> 2, wcol = w & 3;

  const int which = blockIdx.z % 3;
  const int b = blockIdx.z / 3;
  const float* Wm = which == 0 ? Wq : (which == 1 ? Wk : Wv);
  const float* bm = which == 0 ? bq : (which == 1 ? bk : bv);
  unsigned short* Ob = which == 0 ? Qb : (which == 1 ? Kb : Vb);

  const int co_blk = blockIdx.y * 64;
  const int l_blk = blockIdx.x * 128;

  v8f acc[2][2];
#pragma unroll
  for (int i = 0; i < 2; ++i)
#pragma unroll
    for (int n = 0; n < 2; ++n)
#pragma unroll
      for (int j = 0; j < 8; ++j) acc[i][n][j] = 0.0f;

  for (int ci0 = 0; ci0 < C_; ci0 += 32) {
    __syncthreads();
    {  // stage W tile 64x32 (fp32 -> bf16)
      int idx = tid * 8, row = idx >> 5, col = idx & 31;
      const float* g = Wm + (size_t)(co_blk + row) * C_ + ci0 + col;
      float4 f0 = *(const float4*)g;
      float4 f1 = *(const float4*)(g + 4);
      unsigned* d = (unsigned*)ws + (row * 40 + col) / 2;
      d[0] = pack2(f0.x, f0.y); d[1] = pack2(f0.z, f0.w);
      d[2] = pack2(f1.x, f1.y); d[3] = pack2(f1.z, f1.w);
    }
    {  // stage x tile 32x128, transposed to [l][ci], fp32 -> bf16
      int idx = tid * 16, ci = idx >> 7, lof = idx & 127;
      const float* g = x + ((size_t)b * C_ + ci0 + ci) * L_ + l_blk + lof;
#pragma unroll
      for (int k = 0; k < 4; ++k) {
        float4 f = *(const float4*)(g + k * 4);
        xs[(lof + k * 4 + 0) * 40 + ci] = f2bf(f.x);
        xs[(lof + k * 4 + 1) * 40 + ci] = f2bf(f.y);
        xs[(lof + k * 4 + 2) * 40 + ci] = f2bf(f.z);
        xs[(lof + k * 4 + 3) * 40 + ci] = f2bf(f.w);
      }
    }
    __syncthreads();
    v16bf a0 = load_a(ws + (wrow * 32 + ln) * 40, 0, half);
    v16bf a1 = load_a(ws + (wrow * 32 + 16 + ln) * 40, 0, half);
    v16bf b0 = load_b(xs + (wcol * 32 + ln) * 40, 0, half);
    v16bf b1 = load_b(xs + (wcol * 32 + 16 + ln) * 40, 0, half);
    acc[0][0] = wmma_bf16(a0, b0, acc[0][0]);
    acc[0][1] = wmma_bf16(a0, b1, acc[0][1]);
    acc[1][0] = wmma_bf16(a1, b0, acc[1][0]);
    acc[1][1] = wmma_bf16(a1, b1, acc[1][1]);
  }

  const int co_w = co_blk + wrow * 32;
  const int l_w = l_blk + wcol * 32;
#pragma unroll
  for (int mi = 0; mi < 2; ++mi) {
    float bias8[8];
#pragma unroll
    for (int j = 0; j < 8; ++j) bias8[j] = bm[co_w + mi * 16 + half * 8 + j];
#pragma unroll
    for (int ni = 0; ni < 2; ++ni) {
      v8f o = acc[mi][ni];
#pragma unroll
      for (int j = 0; j < 8; ++j) o[j] += bias8[j];
      const int l = l_w + ni * 16 + ln;
      if (which < 2) {  // [b][l][co], 8 consecutive co per lane -> packed dwords
        const int co0 = co_w + mi * 16 + half * 8;
        unsigned* d = (unsigned*)Ob + (((size_t)b * L_ + l) * C_ + co0) / 2;
        d[0] = pack2(o[0], o[1]); d[1] = pack2(o[2], o[3]);
        d[2] = pack2(o[4], o[5]); d[3] = pack2(o[6], o[7]);
      } else {          // V: [b][co][m]
#pragma unroll
        for (int j = 0; j < 8; ++j) {
          const int co = co_w + mi * 16 + half * 8 + j;
          Ob[((size_t)b * C_ + co) * L_ + l] = f2bf(o[j]);
        }
      }
    }
  }
}

// ============================================================
// Kernel 2: fused flash attention.
// Workgroup = 8 waves, 32-query tile. Waves split the 32x64 score
// block (1 WMMA tile each) and split the 512 output channels
// (64 each) for the PV accumulation. Online softmax via LDS
// float atomics (ds_max_num_f32 / ds_add_f32). K/V/Q tiles staged
// with GLOBAL_LOAD_ASYNC_TO_LDS_B128 when available.
// grid: (L/32, B); block 256; dynamic LDS ~95KB.
// ============================================================
__global__ __launch_bounds__(256)
void attn_kernel(const unsigned short* __restrict__ Qb,
                 const unsigned short* __restrict__ Kb,
                 const unsigned short* __restrict__ Vb,
                 float* __restrict__ out) {
  extern __shared__ char smem[];
  unsigned short* qs = (unsigned short*)smem;                            // 32 x 520
  unsigned short* ks = (unsigned short*)(smem + 33280);                  // 64 x 136
  unsigned short* vs = (unsigned short*)(smem + 33280 + 17408);          // 512 x 40
  unsigned short* ps = (unsigned short*)(smem + 33280 + 17408 + 40960);  // 32 x 72
  float* stats  = (float*)(smem + 96256);
  float* m_run  = stats;         // [32] running row max (raw scores)
  float* l_run  = stats + 32;    // [32] running row sum
  float* nmax   = stats + 64;    // [32] block new max
  float* bsum   = stats + 96;    // [32] block exp-sum
  float* rscale = stats + 128;   // [32] accumulator rescale

  const int tid = threadIdx.x;
  const int lane = tid & 31, w = tid >> 5;
  const int half = lane >> 4, ln = lane & 15;
  const int mi = w >> 2, ni = w & 3;  // this wave's S tile (rows mi*16, cols ni*16)
  const int cw = w * 64;              // this wave's output-channel base

  const int b = blockIdx.y;
  const int l0 = blockIdx.x * 32;

  {  // stage Q tile [32 l][512 c] bf16 (persistent)
    int r = tid >> 3, cb = (tid & 7) * 64;
    stage16(qs + r * 520 + cb, Qb + (((size_t)b * L_ + l0 + r) * C_ + cb), 8);
  }
  if (tid < 32) { m_run[tid] = -3.0e38f; l_run[tid] = 0.0f; }

  v8f o[2][4];
#pragma unroll
  for (int mt = 0; mt < 2; ++mt)
#pragma unroll
    for (int nt = 0; nt < 4; ++nt)
#pragma unroll
      for (int j = 0; j < 8; ++j) o[mt][nt][j] = 0.0f;

  stage_fence();
  __syncthreads();

  const float c1 = 0.044194173824159216f * 1.4426950408889634f;  // scale * log2(e)

  for (int m0 = 0; m0 < L_; m0 += 64) {
    // ---------- scores: S(32x64) = Q^T(32xC) * K(Cx64), K-loop over c ----------
    v8f s;
#pragma unroll
    for (int j = 0; j < 8; ++j) s[j] = 0.0f;

    for (int cc = 0; cc < C_; cc += 128) {
      __syncthreads();
      {  // stage K block [64 m][128 c]
        int mr = tid >> 2, cof = (tid & 3) * 32;
        const unsigned short* gp = Kb + (((size_t)b * L_ + m0 + mr) * C_ + cc + cof);
        stage16(ks + mr * 136 + cof, gp, 4);
        if (m0 + 64 < L_) __builtin_prefetch(gp + (size_t)64 * C_, 0, 1);
      }
      stage_fence();
      __syncthreads();
#pragma unroll
      for (int ksub = 0; ksub < 128; ksub += 32) {
        v16bf a  = load_a(qs + (mi * 16 + ln) * 520, cc + ksub, half);
        v16bf bb = load_b(ks + (ni * 16 + ln) * 136, ksub, half);
        s = wmma_bf16(a, bb, s);
      }
    }

    // ---------- online softmax ----------
    // per-row max over this wave's 16 columns (lanes within a half)
    v8f red = s;
    for (int dx = 1; dx < 16; dx <<= 1)
#pragma unroll
      for (int j = 0; j < 8; ++j) red[j] = fmaxf(red[j], __shfl_xor(red[j], dx, 32));

    __syncthreads();
    if (tid < 32) { nmax[tid] = m_run[tid]; bsum[tid] = 0.0f; }
    __syncthreads();
    if (ln == 0) {
#pragma unroll
      for (int j = 0; j < 8; ++j) atomicMax(&nmax[mi * 16 + half * 8 + j], red[j]);
    }
    __syncthreads();

    float mnew[8], p[8];
#pragma unroll
    for (int j = 0; j < 8; ++j) mnew[j] = nmax[mi * 16 + half * 8 + j];
#pragma unroll
    for (int j = 0; j < 8; ++j) p[j] = __builtin_amdgcn_exp2f((s[j] - mnew[j]) * c1);
#pragma unroll
    for (int j = 0; j < 8; ++j)
      ps[(mi * 16 + half * 8 + j) * 72 + ni * 16 + ln] = f2bf(p[j]);

    float rs[8];
#pragma unroll
    for (int j = 0; j < 8; ++j) rs[j] = p[j];
    for (int dx = 1; dx < 16; dx <<= 1)
#pragma unroll
      for (int j = 0; j < 8; ++j) rs[j] += __shfl_xor(rs[j], dx, 32);
    if (ln == 0) {
#pragma unroll
      for (int j = 0; j < 8; ++j) atomicAdd(&bsum[mi * 16 + half * 8 + j], rs[j]);
    }
    __syncthreads();
    if (tid < 32) {
      float sc = __builtin_amdgcn_exp2f((m_run[tid] - nmax[tid]) * c1);
      rscale[tid] = sc;
      l_run[tid] = l_run[tid] * sc + bsum[tid];
      m_run[tid] = nmax[tid];
    }
    __syncthreads();

    // rescale accumulators
#pragma unroll
    for (int mt = 0; mt < 2; ++mt) {
      float sc8[8];
#pragma unroll
      for (int j = 0; j < 8; ++j) sc8[j] = rscale[mt * 16 + half * 8 + j];
#pragma unroll
      for (int nt = 0; nt < 4; ++nt)
#pragma unroll
        for (int j = 0; j < 8; ++j) o[mt][nt][j] *= sc8[j];
    }

    // ---------- O += P(32x64) * V^T(64x512), staged 32 m at a time ----------
    for (int mc = 0; mc < 64; mc += 32) {
      __syncthreads();
      {  // stage V block [512 c][32 m]
        int r0 = tid * 2;
#pragma unroll
        for (int rr = 0; rr < 2; ++rr)
          stage16(vs + (r0 + rr) * 40,
                  Vb + (((size_t)b * C_ + r0 + rr) * L_ + m0 + mc), 4);
      }
      stage_fence();
      __syncthreads();
      v16bf vb[4];
#pragma unroll
      for (int nt = 0; nt < 4; ++nt)
        vb[nt] = load_b(vs + (cw + nt * 16 + ln) * 40, 0, half);
#pragma unroll
      for (int mt = 0; mt < 2; ++mt) {
        v16bf a = load_a(ps + (mt * 16 + ln) * 72, mc, half);
#pragma unroll
        for (int nt = 0; nt < 4; ++nt) o[mt][nt] = wmma_bf16(a, vb[nt], o[mt][nt]);
      }
    }
  }

  // ---------- epilogue: divide by row sums, write out[b][c][l] ----------
  __syncthreads();
#pragma unroll
  for (int mt = 0; mt < 2; ++mt) {
    float inv8[8];
#pragma unroll
    for (int j = 0; j < 8; ++j) inv8[j] = 1.0f / l_run[mt * 16 + half * 8 + j];
#pragma unroll
    for (int nt = 0; nt < 4; ++nt) {
      const int c = cw + nt * 16 + ln;
      float* d = out + ((size_t)b * C_ + c) * L_ + l0 + mt * 16 + half * 8;
#pragma unroll
      for (int j = 0; j < 8; ++j) d[j] = o[mt][nt][j] * inv8[j];
    }
  }
}

// ============================================================
extern "C" void kernel_launch(void* const* d_in, const int* in_sizes, int n_in,
                              void* d_out, int out_size, void* d_ws, size_t ws_size,
                              hipStream_t stream) {
  (void)in_sizes; (void)n_in; (void)out_size; (void)ws_size;
  const float* x  = (const float*)d_in[0];
  const float* Wq = (const float*)d_in[1];
  const float* bq = (const float*)d_in[2];
  const float* Wk = (const float*)d_in[3];
  const float* bk = (const float*)d_in[4];
  const float* Wv = (const float*)d_in[5];
  const float* bv = (const float*)d_in[6];
  float* out = (float*)d_out;

  // bf16 Q/K/V in workspace: 3 * B*L*C * 2B = 48 MB
  unsigned short* Qb = (unsigned short*)d_ws;
  unsigned short* Kb = Qb + (size_t)B_ * L_ * C_;
  unsigned short* Vb = Kb + (size_t)B_ * L_ * C_;

  dim3 g1(L_ / 128, C_ / 64, B_ * 3);
  hipLaunchKernelGGL(qkv_proj_kernel, g1, dim3(256), 0, stream,
                     x, Wq, bq, Wk, bk, Wv, bv, Qb, Kb, Vb);

  const size_t smem = 33280 + 17408 + 40960 + 4608 + 5 * 32 * 4;  // 96896 B
  dim3 g2(L_ / 32, B_);
  hipLaunchKernelGGL(attn_kernel, g2, dim3(256), smem, stream, Qb, Kb, Vb, out);
}
